// QuantizedNN_13142599925839
// MI455X (gfx1250) — 8-bit matrix operand order repaired and checked by reading, compiled, not run on hardware
//
#include <hip/hip_runtime.h>
#include <hip/hip_bf16.h>

typedef __attribute__((ext_vector_type(8))) int v8i;

#define BATCH   16384
#define INDIM   784
#define HIDDEN  4096
#define OUTDIM  10
#define K1P     832
#define N3P     16

#define NSLOTS 8

__global__ void init_hdr_kernel(unsigned* hdr) {
    if (threadIdx.x < NSLOTS) hdr[threadIdx.x] = 0u;
}

__global__ __launch_bounds__(256) void absmax_kernel(
    const float* __restrict__ src, long long n, unsigned* __restrict__ slot)
{
    float m = 0.0f;
    for (long long i = (long long)blockIdx.x * blockDim.x + threadIdx.x;
         i < n; i += (long long)gridDim.x * blockDim.x)
        m = fmaxf(m, fabsf(src[i]));
    #pragma unroll
    for (int o = 16; o; o >>= 1)
        m = fmaxf(m, __shfl_xor(m, o, 32));
    if ((threadIdx.x & 31) == 0)
        atomicMax(slot, __float_as_uint(m));
}

__global__ __launch_bounds__(256) void quant_pack_kernel(
    unsigned char* __restrict__ dst, const float* __restrict__ src,
    int Rsrc, int Ksrc, int Kp, int Rpad,
    const unsigned* __restrict__ hdr, int slot, float nLevels, int modeB)
{
    const int t       = blockIdx.x * blockDim.x + threadIdx.x;
    const int kTiles  = Kp >> 6;
    const int nTiles  = (Rpad >> 4) * kTiles;
    const int tileIdx = t >> 5;
    if (tileIdx >= nTiles) return;
    const int lane = t & 31;
    const int rt   = tileIdx / kTiles;
    const int kt   = tileIdx % kTiles;

    const float mx    = __uint_as_float(hdr[slot]);
    const float scale = fmaxf(mx / nLevels, 1e-8f);
    const float inv   = 1.0f / scale;

    const int half  = lane >> 4;
    const int sub16 = lane & 15;
    v8i outv;
    #pragma unroll
    for (int v = 0; v < 8; ++v) {
        unsigned d = 0;
        #pragma unroll
        for (int c = 0; c < 4; ++c) {
            int k;
            if (!modeB) {
                const int grp = v >> 1, sub = v & 1;
                k = kt * 64 + (grp >> 1) * 32 + half * 16 + (grp & 1) * 8 + sub * 4 + c;
            } else {
                const int big = v >> 2;
                k = kt * 64 + big * 32 + half * 16 + (v & 3) * 4 + c;
            }
            const int row = rt * 16 + sub16;
            float x = 0.0f;
            if (row < Rsrc && k < Ksrc) x = src[(size_t)row * Ksrc + k];
            float q = fminf(fmaxf(rintf(x * inv), -nLevels), nLevels);
            const int qi = (int)q;
            d |= ((unsigned)(qi & 0xFF)) << (8 * c);
        }
        outv[v] = (int)d;
    }
    *(v8i*)(dst + (size_t)tileIdx * 1024 + (size_t)lane * 32) = outv;
}

template<int NT>
__global__ __launch_bounds__(256) void gemm_iu8_kernel(
    const unsigned char* __restrict__ Ap,
    const unsigned char* __restrict__ Bp,
    const float* __restrict__ bias,
    float* __restrict__ out,
    int M, int Npad, int Nout, int Kp,
    unsigned* __restrict__ hdr,
    int slotA, int slotB, float nA, float nB,
    int doRelu, int slotHmax)
{
    const int lane    = threadIdx.x & 31;
    const int wave    = blockIdx.x * (blockDim.x >> 5) + (threadIdx.x >> 5);
    const int nTilesN = Npad / (16 * NT);
    const int mblk    = wave / nTilesN;
    const int nblk    = wave % nTilesN;
    if (mblk * 32 >= M) return;

    const int    kTiles  = Kp >> 6;
    const size_t laneOff = (size_t)lane * 32;
    const int    mt0     = mblk * 2;

    const size_t aOff0 = (size_t)mt0 * kTiles * 1024 + laneOff;
    const size_t aOff1 = aOff0 + (size_t)kTiles * 1024;
    size_t bOff[NT];
    #pragma unroll
    for (int j = 0; j < NT; ++j)
        bOff[j] = (size_t)(nblk * NT + j) * kTiles * 1024 + laneOff;

    v8i acc[2][NT];
    #pragma unroll
    for (int mi = 0; mi < 2; ++mi)
        #pragma unroll
        for (int j = 0; j < NT; ++j)
            acc[mi][j] = (v8i){0, 0, 0, 0, 0, 0, 0, 0};

    v8i a0c = *(const v8i*)(Ap + aOff0);
    v8i a1c = *(const v8i*)(Ap + aOff1);
    v8i bc[NT];
    #pragma unroll
    for (int j = 0; j < NT; ++j) bc[j] = *(const v8i*)(Bp + bOff[j]);

    for (int kt = 1; kt < kTiles; ++kt) {
        const size_t ko = (size_t)kt * 1024;
        const v8i a0n = *(const v8i*)(Ap + aOff0 + ko);
        const v8i a1n = *(const v8i*)(Ap + aOff1 + ko);
        v8i bn[NT];
        #pragma unroll
        for (int j = 0; j < NT; ++j)
            bn[j] = *(const v8i*)(Bp + bOff[j] + ko);
        __builtin_prefetch(Ap + aOff0 + ko + 1024, 0, 3);
        __builtin_prefetch(Bp + bOff[0] + ko + 1024, 0, 3);

        #pragma unroll
        for (int j = 0; j < NT; ++j) {
            acc[0][j] = __builtin_amdgcn_wmma_i32_16x16x64_iu8(
                true, a0c, true, bc[j], acc[0][j], false, false);
            acc[1][j] = __builtin_amdgcn_wmma_i32_16x16x64_iu8(
                true, a1c, true, bc[j], acc[1][j], false, false);
        }
        a0c = a0n; a1c = a1n;
        #pragma unroll
        for (int j = 0; j < NT; ++j) bc[j] = bn[j];
    }
    #pragma unroll
    for (int j = 0; j < NT; ++j) {
        acc[0][j] = __builtin_amdgcn_wmma_i32_16x16x64_iu8(
            true, a0c, true, bc[j], acc[0][j], false, false);
        acc[1][j] = __builtin_amdgcn_wmma_i32_16x16x64_iu8(
            true, a1c, true, bc[j], acc[1][j], false, false);
    }

    const float sA = fmaxf(__uint_as_float(hdr[slotA]) / nA, 1e-8f);
    const float sB = fmaxf(__uint_as_float(hdr[slotB]) / nB, 1e-8f);
    const float s  = sA * sB;

    const int half = lane >> 4;
    const int nIdx = lane & 15;
    float wmax = 0.0f;
    #pragma unroll
    for (int mi = 0; mi < 2; ++mi) {
        #pragma unroll
        for (int j = 0; j < NT; ++j) {
            const int col = (nblk * NT + j) * 16 + nIdx;
            #pragma unroll
            for (int r = 0; r < 8; ++r) {
                const int row = (mt0 + mi) * 16 + r + 8 * half;
                if (col < Nout && row < M) {
                    float v = (float)acc[mi][j][r] * s + bias[col];
                    if (doRelu) v = fmaxf(v, 0.0f);
                    wmax = fmaxf(wmax, fabsf(v));
                    out[(size_t)row * Nout + col] = v;
                }
            }
        }
    }

    if (slotHmax >= 0) {
        #pragma unroll
        for (int o = 16; o; o >>= 1)
            wmax = fmaxf(wmax, __shfl_xor(wmax, o, 32));
        if (lane == 0)
            atomicMax(hdr + slotHmax, __float_as_uint(wmax));
    }
}

static inline int packGrid(int rPad, int kPad) {
    const int threads = (rPad >> 4) * (kPad >> 6) * 32;
    return (threads + 255) / 256;
}

extern "C" void kernel_launch(void* const* d_in, const int* in_sizes, int n_in,
                              void* d_out, int out_size, void* d_ws, size_t ws_size,
                              hipStream_t stream) {
    (void)in_sizes; (void)n_in; (void)out_size; (void)ws_size;
    const float* x  = (const float*)d_in[0];
    const float* W1 = (const float*)d_in[1];
    const float* b1 = (const float*)d_in[2];
    const float* W2 = (const float*)d_in[3];
    const float* b2 = (const float*)d_in[4];
    const float* W3 = (const float*)d_in[5];
    const float* b3 = (const float*)d_in[6];
    float* out = (float*)d_out;

    unsigned char* ws  = (unsigned char*)d_ws;
    unsigned*      hdr = (unsigned*)ws;

    const size_t off_wp1 = 256;
    const size_t off_wp2 = off_wp1 + (size_t)HIDDEN * K1P;
    const size_t off_wp3 = off_wp2 + (size_t)HIDDEN * HIDDEN;
    const size_t off_ap1 = off_wp3 + (size_t)N3P * HIDDEN;
    const size_t off_ap2 = off_ap1 + (size_t)BATCH * K1P;
    const size_t off_h   = off_ap2 + (size_t)BATCH * HIDDEN;
    unsigned char* wp1 = ws + off_wp1;
    unsigned char* wp2 = ws + off_wp2;
    unsigned char* wp3 = ws + off_wp3;
    unsigned char* ap1 = ws + off_ap1;
    unsigned char* ap2 = ws + off_ap2;
    float*         H   = (float*)(ws + off_h);

    init_hdr_kernel<<<1, 32, 0, stream>>>(hdr);

    absmax_kernel<<<1024, 256, 0, stream>>>(x,  (long long)BATCH * INDIM,   hdr + 0);
    absmax_kernel<<<1024, 256, 0, stream>>>(W1, (long long)HIDDEN * INDIM,  hdr + 1);
    absmax_kernel<<<2048, 256, 0, stream>>>(W2, (long long)HIDDEN * HIDDEN, hdr + 2);
    absmax_kernel<<<64,   256, 0, stream>>>(W3, (long long)OUTDIM * HIDDEN, hdr + 3);

    quant_pack_kernel<<<packGrid(HIDDEN, K1P), 256, 0, stream>>>(
        wp1, W1, HIDDEN, INDIM, K1P, HIDDEN, hdr, 1, 127.0f, 1);
    quant_pack_kernel<<<packGrid(HIDDEN, HIDDEN), 256, 0, stream>>>(
        wp2, W2, HIDDEN, HIDDEN, HIDDEN, HIDDEN, hdr, 2, 127.0f, 1);
    quant_pack_kernel<<<packGrid(N3P, HIDDEN), 256, 0, stream>>>(
        wp3, W3, OUTDIM, HIDDEN, HIDDEN, N3P, hdr, 3, 127.0f, 1);
    quant_pack_kernel<<<packGrid(BATCH, K1P), 256, 0, stream>>>(
        ap1, x, BATCH, INDIM, K1P, BATCH, hdr, 0, 7.0f, 0);

    {
        const int waves = (BATCH / 32) * (HIDDEN / 64);
        gemm_iu8_kernel<4><<<waves / 8, 256, 0, stream>>>(
            ap1, wp1, b1, H, BATCH, HIDDEN, HIDDEN, K1P,
            hdr, 0, 1, 7.0f, 127.0f, 1, 4);
    }
    quant_pack_kernel<<<packGrid(BATCH, HIDDEN), 256, 0, stream>>>(
        ap2, H, BATCH, HIDDEN, HIDDEN, BATCH, hdr, 4, 7.0f, 0);

    {
        const int waves = (BATCH / 32) * (HIDDEN / 64);
        gemm_iu8_kernel<4><<<waves / 8, 256, 0, stream>>>(
            ap2, wp2, b2, H, BATCH, HIDDEN, HIDDEN, HIDDEN,
            hdr, 4, 2, 7.0f, 127.0f, 1, 5);
    }
    quant_pack_kernel<<<packGrid(BATCH, HIDDEN), 256, 0, stream>>>(
        ap2, H, BATCH, HIDDEN, HIDDEN, BATCH, hdr, 5, 7.0f, 0);

    {
        const int waves = (BATCH / 32) * (N3P / 16);
        gemm_iu8_kernel<1><<<waves / 8, 256, 0, stream>>>(
            ap2, wp3, b3, out, BATCH, N3P, OUTDIM, HIDDEN,
            hdr, 5, 3, 7.0f, 127.0f, 0, -1);
    }
}
